// DbrxExperts__57698590654865
// MI455X (gfx1250) — matrix operand order repaired and checked by reading, not compiled, not run on hardware
//
#include <hip/hip_runtime.h>

// Problem constants (reference: B=4,S=2048,H=1024,F=2048,E=16,K=4)
#define Hc 1024
#define Fc 2048
#define Ec 16
#define Tc 8192

// Tiling
#define TM 128
#define TN 128
#define TKc 32
#define LDSK 40    // padded K-stride (elements) -> 80B rows, 16B aligned
#define TILE_E (TM * LDSK)

// ---- gfx1250 async global->LDS path (guarded; falls back to uint4 copies) ----
#if defined(__has_builtin)
#if __has_builtin(__builtin_amdgcn_global_load_async_to_lds_b128)
#define USE_ASYNC_LDS 1
#endif
#endif
#ifndef USE_ASYNC_LDS
#define USE_ASYNC_LDS 0
#endif

typedef __attribute__((ext_vector_type(16))) __bf16 v16bf;
typedef __attribute__((ext_vector_type(8)))  float  v8f;
typedef __attribute__((__vector_size__(16))) int    v4i32;

union FragU { v16bf v; uint4 q[2]; };

__device__ __forceinline__ unsigned short f2bf(float a) {
  unsigned int u = __float_as_uint(a);
  return (unsigned short)((u + 0x7FFFu + ((u >> 16) & 1u)) >> 16);  // RNE
}
__device__ __forceinline__ unsigned int pack2(float a, float b) {
  return (unsigned int)f2bf(a) | ((unsigned int)f2bf(b) << 16);
}

// Copy 16 bf16 (32B) global -> LDS. Async-tensor path on gfx1250.
__device__ __forceinline__ void copy_a16(const unsigned short* g, unsigned short* l) {
#if USE_ASYNC_LDS
  __attribute__((address_space(1))) v4i32* gsrc =
      (__attribute__((address_space(1))) v4i32*)(unsigned long long)g;
  __attribute__((address_space(3))) v4i32* ldst =
      (__attribute__((address_space(3))) v4i32*)l;
  __builtin_amdgcn_global_load_async_to_lds_b128(gsrc, ldst, 0, 0);
  __builtin_amdgcn_global_load_async_to_lds_b128(gsrc + 1, ldst + 1, 0, 0);
#else
  ((uint4*)l)[0] = ((const uint4*)g)[0];
  ((uint4*)l)[1] = ((const uint4*)g)[1];
#endif
}

__device__ __forceinline__ void wait_async_all() {
#if USE_ASYNC_LDS
#if __has_builtin(__builtin_amdgcn_s_wait_asynccnt)
  __builtin_amdgcn_s_wait_asynccnt(0);
#else
  asm volatile("s_wait_asynccnt 0x0" ::: "memory");
#endif
#endif
}

// Load 16 f32, round to 8 packed bf16x2 regs
__device__ __forceinline__ void load_cvt16(const float* s, unsigned int* d) {
#pragma unroll
  for (int i = 0; i < 4; i++) {
    float4 f = *(const float4*)(s + i * 4);
    d[i * 2 + 0] = pack2(f.x, f.y);
    d[i * 2 + 1] = pack2(f.z, f.w);
  }
}
__device__ __forceinline__ void store16(const unsigned int* r, unsigned short* l) {
  unsigned int* d = (unsigned int*)l;
#pragma unroll
  for (int i = 0; i < 8; i++) d[i] = r[i];
}

// 16-bit A fragment 16x32 (ISA 7.12.2): lane m = l&15, K-half = l>>4;
// per-lane chunks at k = 8*kh .. and k = 16+8*kh ..
__device__ __forceinline__ v16bf load_a_frag(const unsigned short* base, int lane) {
  FragU f;
  const unsigned short* p = base + (lane & 15) * LDSK + ((lane >> 4) << 3);
  f.q[0] = *(const uint4*)(p);
  f.q[1] = *(const uint4*)(p + 16);
  return f.v;
}
// 16-bit B fragment 32x16: lane col n = l&15, K chunks at 8*(l>>4) and 16+8*(l>>4)
__device__ __forceinline__ v16bf load_b_frag(const unsigned short* base, int lane) {
  FragU f;
  const unsigned short* p = base + (lane & 15) * LDSK + ((lane >> 4) << 3);
  f.q[0] = *(const uint4*)(p);
  f.q[1] = *(const uint4*)(p + 16);
  return f.v;
}

__device__ __forceinline__ v8f wmma_bf16(v16bf a, v16bf b, v8f c) {
  return __builtin_amdgcn_wmma_f32_16x16x32_bf16(false, a, false, b, (short)0, c,
                                                 false, false);
}

// ---------------- x (f32) -> bf16 ----------------
__global__ void __launch_bounds__(256) cvt_x_kernel(const float* __restrict__ x,
                                                    unsigned short* __restrict__ xb,
                                                    int n) {
  int i = (blockIdx.x * 256 + threadIdx.x) * 4;
  if (i >= n) return;
  float4 f = *(const float4*)(x + i);
  uint2 p;
  p.x = pack2(f.x, f.y);
  p.y = pack2(f.z, f.w);
  *(uint2*)(xb + i) = p;
}

// ------- cw[e][t] = sum_k (top_experts[t,k]==e) * top_weights[t,k] -------
__global__ void __launch_bounds__(256) cw_kernel(const float* __restrict__ tw,
                                                 const int* __restrict__ te,
                                                 float* __restrict__ cw) {
  int idx = blockIdx.x * 256 + threadIdx.x;  // e*Tc + t, grid covers Ec*Tc
  int e = idx >> 13;                          // / Tc
  int t = idx & (Tc - 1);
  float s = 0.f;
#pragma unroll
  for (int k = 0; k < 4; k++)
    if (te[t * 4 + k] == e) s += tw[t * 4 + k];
  cw[idx] = s;
}

// ---------------- gate/up: act[t,f] = silu(x@w1^T) * (x@v1^T) ----------------
// Double-buffered LDS pipeline, one barrier per K-step; A-tile goes through the
// async global->LDS path, weight tiles are converted f32->bf16 in VALU while
// the previous tile's WMMAs run.
__global__ void __launch_bounds__(256, 2) moe_gateup_kernel(
    const unsigned short* __restrict__ xb,   // [T,H] bf16
    const float* __restrict__ w1e,           // [F,H] f32 (expert slice)
    const float* __restrict__ v1e,           // [F,H] f32
    unsigned short* __restrict__ act) {      // [T,F] bf16
  __shared__ unsigned short As[2 * TILE_E];
  __shared__ unsigned short Bs1[2 * TILE_E];
  __shared__ unsigned short Bs2[2 * TILE_E];

  const int tid = threadIdx.x;
  const int lane = tid & 31;
  const int wid = tid >> 5;
  const int wm = wid & 3;   // 4 waves x 32 rows
  const int wn = wid >> 2;  // 2 waves x 64 cols

  const int t0 = (blockIdx.x >> 4) * TM;  // Fc/TN = 16 tiles in N
  const int f0 = (blockIdx.x & 15) * TN;

  v8f zero8;
#pragma unroll
  for (int q = 0; q < 8; q++) zero8[q] = 0.f;
  v8f accg[2][4], accu[2][4];
#pragma unroll
  for (int i = 0; i < 2; i++)
#pragma unroll
    for (int j = 0; j < 4; j++) { accg[i][j] = zero8; accu[i][j] = zero8; }

  const int lrow = tid >> 1;   // 0..127
  const int lhalf = tid & 1;   // 16-element half of the 32-wide K slab
  const unsigned short* gA = xb + (size_t)(t0 + lrow) * Hc + lhalf * 16;
  const float* gW1 = w1e + (size_t)(f0 + lrow) * Hc + lhalf * 16;
  const float* gWv = v1e + (size_t)(f0 + lrow) * Hc + lhalf * 16;
  const int loff = lrow * LDSK + lhalf * 16;

  unsigned int bst1[8], bst2[8];

  // prologue: tile kk=0 -> buffer 0
  copy_a16(gA, As + loff);
  load_cvt16(gW1, bst1);
  load_cvt16(gWv, bst2);
  store16(bst1, Bs1 + loff);
  store16(bst2, Bs2 + loff);
  wait_async_all();
  __syncthreads();

  for (int kk = 0; kk < Hc; kk += TKc) {
    const int cur = (kk >> 5) & 1;
    const int nxt = cur ^ 1;
    const bool more = (kk + TKc) < Hc;

    if (more) {  // stage next tile while current computes
      copy_a16(gA + kk + TKc, As + nxt * TILE_E + loff);           // async
      load_cvt16(gW1 + kk + TKc, bst1);
      load_cvt16(gWv + kk + TKc, bst2);
      if (kk + 2 * TKc < Hc) {
        __builtin_prefetch(gW1 + kk + 2 * TKc, 0, 1);
        __builtin_prefetch(gWv + kk + 2 * TKc, 0, 1);
      }
    }

    const unsigned short* Ac = As + cur * TILE_E;
    const unsigned short* B1c = Bs1 + cur * TILE_E;
    const unsigned short* B2c = Bs2 + cur * TILE_E;
    v16bf afrag[2];
#pragma unroll
    for (int im = 0; im < 2; im++)
      afrag[im] = load_a_frag(Ac + (wm * 32 + im * 16) * LDSK, lane);
#pragma unroll
    for (int in = 0; in < 4; in++) {
      v16bf b1 = load_b_frag(B1c + (wn * 64 + in * 16) * LDSK, lane);
      accg[0][in] = wmma_bf16(afrag[0], b1, accg[0][in]);
      accg[1][in] = wmma_bf16(afrag[1], b1, accg[1][in]);
      v16bf b2 = load_b_frag(B2c + (wn * 64 + in * 16) * LDSK, lane);
      accu[0][in] = wmma_bf16(afrag[0], b2, accu[0][in]);
      accu[1][in] = wmma_bf16(afrag[1], b2, accu[1][in]);
    }

    if (more) {
      store16(bst1, Bs1 + nxt * TILE_E + loff);
      store16(bst2, Bs2 + nxt * TILE_E + loff);
      wait_async_all();
      __syncthreads();
    }
  }

  // epilogue: silu(gate)*up -> bf16 act.  C/D layout: VGPR r, lanes0-15 M=r,
  // lanes16-31 M=8+r, N=lane&15
  const int mbase = t0 + wm * 32 + ((lane >> 4) << 3);
  const int nbase = f0 + wn * 64 + (lane & 15);
#pragma unroll
  for (int im = 0; im < 2; im++)
#pragma unroll
    for (int in = 0; in < 4; in++)
#pragma unroll
      for (int r = 0; r < 8; r++) {
        float g = accg[im][in][r];
        float u = accu[im][in][r];
        float s = g / (1.f + __expf(-g));
        act[(size_t)(mbase + im * 16 + r) * Fc + nbase + in * 16] = f2bf(s * u);
      }
}

// ---------------- down: out[t,h] (+)= (act @ w2) * cw[e][t] ----------------
__global__ void __launch_bounds__(256, 2) moe_down_kernel(
    const unsigned short* __restrict__ act,  // [T,F] bf16
    const float* __restrict__ w2e,           // [F,H] f32
    const float* __restrict__ cwe,           // [T]
    float* __restrict__ out,                 // [T,H]
    int accumulate) {
  __shared__ unsigned short AsD[2 * TILE_E];
  __shared__ unsigned short BsD[2 * TILE_E];  // transposed: [n=h][k=f]

  const int tid = threadIdx.x;
  const int lane = tid & 31;
  const int wid = tid >> 5;
  const int wm = wid & 3;
  const int wn = wid >> 2;

  const int t0 = (blockIdx.x >> 3) * TM;  // Hc/TN = 8 tiles in N
  const int h0 = (blockIdx.x & 7) * TN;

  v8f zero8;
#pragma unroll
  for (int q = 0; q < 8; q++) zero8[q] = 0.f;
  v8f acc[2][4];
#pragma unroll
  for (int i = 0; i < 2; i++)
#pragma unroll
    for (int j = 0; j < 4; j++) acc[i][j] = zero8;

  const int lrow = tid >> 1;
  const int lhalf = tid & 1;
  const int fr = tid >> 3;  // 0..31: w2 row (k) within slab
  const int c8 = tid & 7;   // 8 groups of 16 h columns
  const unsigned short* gA = act + (size_t)(t0 + lrow) * Fc + lhalf * 16;
  const float* gW = w2e + (size_t)fr * Hc + h0 + c8 * 16;
  const int loff = lrow * LDSK + lhalf * 16;

  float4 bst[4];

  // prologue: tile kk=0 -> buffer 0
  copy_a16(gA, AsD + loff);
#pragma unroll
  for (int i = 0; i < 4; i++) bst[i] = *(const float4*)(gW + i * 4);
#pragma unroll
  for (int i = 0; i < 4; i++) {
    int n = c8 * 16 + i * 4;
    BsD[(n + 0) * LDSK + fr] = f2bf(bst[i].x);
    BsD[(n + 1) * LDSK + fr] = f2bf(bst[i].y);
    BsD[(n + 2) * LDSK + fr] = f2bf(bst[i].z);
    BsD[(n + 3) * LDSK + fr] = f2bf(bst[i].w);
  }
  wait_async_all();
  __syncthreads();

  for (int kk = 0; kk < Fc; kk += TKc) {
    const int cur = (kk >> 5) & 1;
    const int nxt = cur ^ 1;
    const bool more = (kk + TKc) < Fc;

    if (more) {
      copy_a16(gA + kk + TKc, AsD + nxt * TILE_E + loff);  // async
      const float* s = gW + (size_t)(kk + TKc) * Hc;
#pragma unroll
      for (int i = 0; i < 4; i++) bst[i] = *(const float4*)(s + i * 4);
      if (kk + 2 * TKc < Fc)
        __builtin_prefetch(gW + (size_t)(kk + 2 * TKc) * Hc, 0, 1);
    }

    const unsigned short* Ac = AsD + cur * TILE_E;
    const unsigned short* Bc = BsD + cur * TILE_E;
    v16bf afrag[2];
#pragma unroll
    for (int im = 0; im < 2; im++)
      afrag[im] = load_a_frag(Ac + (wm * 32 + im * 16) * LDSK, lane);
#pragma unroll
    for (int in = 0; in < 4; in++) {
      v16bf b = load_b_frag(Bc + (wn * 64 + in * 16) * LDSK, lane);
      acc[0][in] = wmma_bf16(afrag[0], b, acc[0][in]);
      acc[1][in] = wmma_bf16(afrag[1], b, acc[1][in]);
    }

    if (more) {
      unsigned short* Bn = BsD + nxt * TILE_E;
#pragma unroll
      for (int i = 0; i < 4; i++) {
        int n = c8 * 16 + i * 4;
        Bn[(n + 0) * LDSK + fr] = f2bf(bst[i].x);
        Bn[(n + 1) * LDSK + fr] = f2bf(bst[i].y);
        Bn[(n + 2) * LDSK + fr] = f2bf(bst[i].z);
        Bn[(n + 3) * LDSK + fr] = f2bf(bst[i].w);
      }
      wait_async_all();
      __syncthreads();
    }
  }

  const int mloc0 = ((lane >> 4) << 3);
  const int nloc = lane & 15;
#pragma unroll
  for (int im = 0; im < 2; im++)
#pragma unroll
    for (int in = 0; in < 4; in++)
#pragma unroll
      for (int r = 0; r < 8; r++) {
        int t = t0 + wm * 32 + im * 16 + mloc0 + r;
        int h = h0 + wn * 64 + in * 16 + nloc;
        float v = acc[im][in][r] * cwe[t];
        size_t o = (size_t)t * Hc + h;
        if (accumulate) out[o] += v; else out[o] = v;
      }
}

extern "C" void kernel_launch(void* const* d_in, const int* in_sizes, int n_in,
                              void* d_out, int out_size, void* d_ws, size_t ws_size,
                              hipStream_t stream) {
  (void)in_sizes; (void)n_in; (void)out_size; (void)ws_size;
  const float* x = (const float*)d_in[0];
  // d_in[1] = router probs, unused in fwd
  const float* top_w = (const float*)d_in[2];
  const int* top_e = (const int*)d_in[3];
  const float* w1 = (const float*)d_in[4];
  const float* v1 = (const float*)d_in[5];
  const float* w2 = (const float*)d_in[6];
  float* out = (float*)d_out;

  // Workspace layout: xb (T*H bf16), act (T*F bf16), cw (E*T f32)
  char* ws = (char*)d_ws;
  const size_t XB_BYTES = (size_t)Tc * Hc * 2;   // 16 MB
  const size_t ACT_BYTES = (size_t)Tc * Fc * 2;  // 32 MB
  unsigned short* xb = (unsigned short*)ws;
  unsigned short* act = (unsigned short*)(ws + XB_BYTES);
  float* cw = (float*)(ws + XB_BYTES + ACT_BYTES);

  const int nXH = Tc * Hc;
  cvt_x_kernel<<<nXH / 1024, 256, 0, stream>>>(x, xb, nXH);
  cw_kernel<<<(Ec * Tc) / 256, 256, 0, stream>>>(top_w, top_e, cw);

  const size_t WSTRIDE = (size_t)Fc * Hc;    // per-expert weight slice
  const int gridGU = (Tc / TM) * (Fc / TN);  // 64 * 16 = 1024
  const int gridDN = (Tc / TM) * (Hc / TN);  // 64 * 8  = 512
  for (int e = 0; e < Ec; e++) {
    moe_gateup_kernel<<<gridGU, 256, 0, stream>>>(xb, w1 + (size_t)e * WSTRIDE,
                                                  v1 + (size_t)e * WSTRIDE, act);
    moe_down_kernel<<<gridDN, 256, 0, stream>>>(act, w2 + (size_t)e * WSTRIDE,
                                                cw + (size_t)e * Tc, out,
                                                e == 0 ? 0 : 1);
  }
}
